// RefIndexer_88235808129232
// MI455X (gfx1250) — matrix operand order repaired and checked by reading, not compiled, not run on hardware
//
#include <hip/hip_runtime.h>
#include <hip/hip_bf16.h>

// ---------------------------------------------------------------------------
// MI455X (gfx1250, wave32) implementation of the MLA "lightning indexer".
// All GEMMs run through v_wmma_f32_16x16x32_bf16. Data staged as bf16 in ws.
// The scores kernel stages its 64KB q-tile via the Tensor Data Mover
// (tensor_load_to_lds + s_wait_tensorcnt) instead of VGPR round-trips.
// ---------------------------------------------------------------------------

typedef __attribute__((ext_vector_type(16))) __bf16 v16bf;
typedef __attribute__((ext_vector_type(8)))  __bf16 v8bf;
typedef __attribute__((ext_vector_type(8)))  float  v8f;
typedef unsigned int u32;
typedef __attribute__((ext_vector_type(4))) u32 u32x4;
typedef __attribute__((ext_vector_type(8))) u32 u32x8;

constexpr int   kS     = 4096;
constexpr int   kDIM   = 3072;
constexpr int   kQLORA = 1536;
constexpr int   kH     = 16;
constexpr int   kD     = 128;
constexpr int   kTOPK  = 2048;
constexpr float kMIN   = -3.4028234663852886e38f;
// H^-0.5 * D^-0.5
constexpr float kWSCALE = 0.022097086912079608f;

// ---------------- WMMA fragment helpers (contiguous k order: k = 16*half + s) ----------------

__device__ __forceinline__ v16bf concat8(v8bf lo, v8bf hi) {
  v16bf r;
#pragma unroll
  for (int i = 0; i < 8; ++i) { r[i] = lo[i]; r[i + 8] = hi[i]; }
  return r;
}

// A 16x32 bf16 (MxK), A row-major [M][K]:
// lane<16 : row=lane,    K = 0..15
// lane>=16: row=lane-16, K = 16..31
__device__ __forceinline__ v16bf load_a_frag(const __bf16* A, int lda,
                                             int row0, int k0, int lane) {
  const int m = lane & 15, half = lane >> 4;
  const __bf16* p = A + (long)(row0 + m) * lda + k0 + half * 16;
  v8bf lo = *(const v8bf*)p;
  v8bf hi = *(const v8bf*)(p + 8);
  return concat8(lo, hi);
}

// B 32x16 bf16 (KxN), stored transposed [N][K] (each logical column contiguous):
// lane<16 : col=lane,    K = 0..15 ; lane>=16: col=lane-16, K = 16..31
__device__ __forceinline__ v16bf load_b_frag(const __bf16* BT, int ldb,
                                             int n0, int k0, int lane) {
  const int n = lane & 15, kb = (lane >> 4) * 16;
  return *(const v16bf*)(BT + (long)(n0 + n) * ldb + k0 + kb);
}

// ---------------- k0: fp32 -> bf16 convert / transpose-convert --------------

__global__ void __launch_bounds__(256) cvt_bf16_kernel(const float* __restrict__ src,
                                                       __bf16* __restrict__ dst, int n) {
  int i = blockIdx.x * 256 + threadIdx.x;
  if (i < n) dst[i] = (__bf16)src[i];
}

// dst[N][K] = (bf16) src[K][N]
__global__ void transpose_cvt_kernel(const float* __restrict__ src,
                                     __bf16* __restrict__ dst, int K, int N) {
  __shared__ float tile[32][33];
  const int kb = blockIdx.y * 32, nb = blockIdx.x * 32;
  const int tx = threadIdx.x, ty = threadIdx.y;
#pragma unroll
  for (int r = 0; r < 32; r += 8) {
    int k = kb + ty + r, n = nb + tx;
    tile[ty + r][tx] = (k < K && n < N) ? src[(long)k * N + n] : 0.0f;
  }
  __syncthreads();
#pragma unroll
  for (int r = 0; r < 32; r += 8) {
    int n = nb + ty + r, k = kb + tx;
    if (n < N && k < K) dst[(long)n * K + k] = (__bf16)tile[tx][ty + r];
  }
}

// ---------------- k1: q = qr @ wq_b, RoPE on first 64 dims ------------------
// grid = (S/16, H); block = 256 (8 waves; wave w -> cols head*128 + w*16)

__global__ void __launch_bounds__(256) qproj_kernel(const __bf16* __restrict__ qrb,
                                                    const __bf16* __restrict__ wqbT,
                                                    const float* __restrict__ cosp,
                                                    const float* __restrict__ sinp,
                                                    __bf16* __restrict__ qf) {
  const int lane = threadIdx.x & 31, wave = threadIdx.x >> 5;
  const int row0 = blockIdx.x * 16;
  const int head = blockIdx.y;
  const int n0 = head * kD + wave * 16;

  v8f acc = {};
  for (int k0 = 0; k0 < kQLORA; k0 += 32) {
    v16bf a = load_a_frag(qrb, kQLORA, row0, k0, lane);
    v16bf b = load_b_frag(wqbT, kQLORA, n0, k0, lane);
    acc = __builtin_amdgcn_wmma_f32_16x16x32_bf16(false, a, false, b,
                                                  (short)0, acc, false, false);
  }

  __shared__ float tile[16][128];
  {
    const int col = lane & 15, halfm = (lane >> 4) * 8;
#pragma unroll
    for (int r = 0; r < 8; ++r) tile[halfm + r][wave * 16 + col] = acc[r];
  }
  __syncthreads();

  // RoPE epilogue: out[j<32] = t[2j]c - t[2j+1]s ; out[32+j] = t[2j+1]c + t[2j]s
#pragma unroll
  for (int e = 0; e < 8; ++e) {
    int id = threadIdx.x + e * 256;
    int row = id >> 7, col = id & 127;
    int sa = row0 + row;
    float v;
    if (col < 32) {
      float c = cosp[sa * 64 + col], s = sinp[sa * 64 + col];
      v = tile[row][2 * col] * c - tile[row][2 * col + 1] * s;
    } else if (col < 64) {
      int j = col - 32;
      float c = cosp[sa * 64 + col], s = sinp[sa * 64 + col];
      v = tile[row][2 * j + 1] * c + tile[row][2 * j] * s;
    } else {
      v = tile[row][col];
    }
    qf[((long)head * kS + sa) * kD + col] = (__bf16)v;
  }
}

// ---------------- k2: k = rmsnorm(x@wk)+RoPE ; weights = x@ww ---------------
// grid = S/16 ; block = 288 (9 waves: 8 for the 128 k-cols, 1 for 16 weights)

__global__ void __launch_bounds__(288) kproj_kernel(const __bf16* __restrict__ xb,
                                                    const __bf16* __restrict__ wkT,
                                                    const __bf16* __restrict__ wwT,
                                                    const float* __restrict__ cosp,
                                                    const float* __restrict__ sinp,
                                                    const float* __restrict__ knw,
                                                    __bf16* __restrict__ kf,
                                                    float* __restrict__ wts) {
  const int lane = threadIdx.x & 31, wave = threadIdx.x >> 5;
  const int row0 = blockIdx.x * 16;

  __shared__ float tile[16][144];
  __shared__ float rrms[16];

  const __bf16* BT = (wave < 8) ? wkT : wwT;
  const int n0 = (wave < 8) ? wave * 16 : 0;

  v8f acc = {};
  for (int k0 = 0; k0 < kDIM; k0 += 32) {
    v16bf a = load_a_frag(xb, kDIM, row0, k0, lane);
    v16bf b = load_b_frag(BT, kDIM, n0, k0, lane);
    acc = __builtin_amdgcn_wmma_f32_16x16x32_bf16(false, a, false, b,
                                                  (short)0, acc, false, false);
  }
  {
    const int col = lane & 15, halfm = (lane >> 4) * 8;
    const int cbase = (wave < 8) ? wave * 16 : 128;
#pragma unroll
    for (int r = 0; r < 8; ++r) tile[halfm + r][cbase + col] = acc[r];
  }
  __syncthreads();

  if (threadIdx.x < 16) {
    float ss = 0.0f;
    for (int d = 0; d < kD; ++d) { float v = tile[threadIdx.x][d]; ss += v * v; }
    rrms[threadIdx.x] = rsqrtf(ss * (1.0f / kD) + 1e-6f);
  }
  __syncthreads();

#pragma unroll
  for (int e = 0; e < 8; ++e) {
    int id = threadIdx.x + e * 288;
    int row = id / 144, col = id % 144;
    int sa = row0 + row;
    if (col < 128) {
      float v;
      if (col < 64) {
        float c = cosp[sa * 64 + col], s = sinp[sa * 64 + col];
        if (col < 32) {
          float a0 = tile[row][2 * col] * rrms[row] * knw[2 * col];
          float a1 = tile[row][2 * col + 1] * rrms[row] * knw[2 * col + 1];
          v = a0 * c - a1 * s;
        } else {
          int j = col - 32;
          float a0 = tile[row][2 * j] * rrms[row] * knw[2 * j];
          float a1 = tile[row][2 * j + 1] * rrms[row] * knw[2 * j + 1];
          v = a1 * c + a0 * s;
        }
      } else {
        v = tile[row][col] * rrms[row] * knw[col];
      }
      kf[(long)sa * kD + col] = (__bf16)v;
    } else {
      int h = col - 128;
      wts[sa * kH + h] = tile[row][col] * kWSCALE;
    }
  }
}

// ---------------- k3: index_score = sum_h w[h,q]*relu(q·kT) + causal --------
// grid = (S/16 qtiles, S/128 kchunks); block = 256 (wave w -> 16-key tile)
// The 64KB q-tile (16 heads x 16 rows x 128 d, bf16) is fetched by the
// Tensor Data Mover: 2D tile, dim0 = 2048 contiguous elements (one head's
// 16x128 rows), dim1 = 16 heads with stride S*D elements.

__global__ void __launch_bounds__(256) scores_kernel(const __bf16* __restrict__ qf,
                                                     const __bf16* __restrict__ kf,
                                                     const float* __restrict__ wts,
                                                     float* __restrict__ score) {
  extern __shared__ unsigned char smem3[];
  __bf16* qs = (__bf16*)smem3;                 // [16 heads][16 rows][128] = 64KB
  float*  wl = (float*)(qs + 16 * 16 * 128);   // [16 rows][16 heads]

  const int row0 = blockIdx.x * 16;
  const int lane = threadIdx.x & 31, wave = threadIdx.x >> 5;
  const int n0 = blockIdx.y * 128 + wave * 16;

  // --- TDM staging: one wave issues the descriptor, TENSORcnt tracks it. ---
  if (threadIdx.x < 32) {
    const unsigned long long ga =
        (unsigned long long)(const void*)(qf + (long)row0 * kD);
    u32x4 g0;
    g0[0] = 1u;                                     // count=1, user descriptor
    g0[1] = (u32)(size_t)qs;                        // lds_addr (flat low bits)
    g0[2] = (u32)ga;                                // global_addr[31:0]
    g0[3] = (u32)((ga >> 32) & 0x01FFFFFFu) | (2u << 30);  // addr[56:32]|type=2
    u32x8 g1;
    g1[0] = 1u << 16;                               // data_size = 2 bytes
    g1[1] = (2048u & 0xFFFFu) << 16;                // tensor_dim0[15:0]
    g1[2] = (2048u >> 16) | (16u << 16);            // tensor_dim0[31:16]|dim1
    g1[3] = (2048u << 16);                          // tile_dim0 = 2048
    g1[4] = 16u;                                    // tile_dim1 = 16, tile_dim2=0
    g1[5] = (u32)(kS * kD);                         // tensor_dim0_stride[31:0]
    g1[6] = 0u;                                     // stride[47:32] | dim1_stride
    g1[7] = 0u;
    asm volatile("tensor_load_to_lds %0, %1" :: "s"(g0), "s"(g1) : "memory");
    __builtin_amdgcn_s_wait_tensorcnt(0);
  }
  wl[threadIdx.x] = wts[row0 * kH + threadIdx.x];   // [row][h] layout
  __syncthreads();

  __builtin_prefetch(kf + (long)(n0 + (lane & 15)) * kD, 0, 3);

  v8f fin = {};
  for (int h = 0; h < kH; ++h) {
    v8f acc = {};
    const int m = lane & 15, half = lane >> 4;
#pragma unroll
    for (int ks = 0; ks < kD; ks += 32) {
      const __bf16* pa = qs + (h * 16 + m) * kD + ks + half * 16;
      v8bf lo = *(const v8bf*)pa;
      v8bf hi = *(const v8bf*)(pa + 8);
      v16bf a = concat8(lo, hi);
      v16bf b = *(const v16bf*)(kf + (long)(n0 + m) * kD + ks + half * 16);
      acc = __builtin_amdgcn_wmma_f32_16x16x32_bf16(false, a, false, b,
                                                    (short)0, acc, false, false);
    }
    const int halfm = (lane >> 4) * 8;
#pragma unroll
    for (int r = 0; r < 8; ++r) {
      float v = acc[r];
      float w = wl[(halfm + r) * kH + h];
      fin[r] += (v > 0.0f) ? v * w : 0.0f;
    }
  }

  {
    const int col = lane & 15, halfm = (lane >> 4) * 8;
#pragma unroll
    for (int r = 0; r < 8; ++r) {
      int qa = row0 + halfm + r;
      int ka = n0 + col;
      float v = fin[r] + ((ka > qa) ? kMIN : 0.0f);
      score[(long)qa * kS + ka] = v;
    }
  }
}

// ---------------- k4: per-row ordered top-k via LDS radix sort --------------
// One block per query row. 8 passes of stable 4-bit LSD radix over 4096
// (key,idx) pairs entirely in LDS (320KB/WGP on CDNA5 makes this trivial).

__global__ void __launch_bounds__(256) topk_kernel(float* __restrict__ score,
                                                   int* __restrict__ topk,
                                                   float* __restrict__ maskout) {
  extern __shared__ unsigned char smem4[];
  u32* keyA = (u32*)smem4;             // 4096
  u32* keyB = keyA + kS;               // 4096
  u32* idxA = keyB + kS;               // 4096
  u32* idxB = idxA + kS;               // 4096
  u32* cnt  = idxB + kS;               // 256*16
  u32* dig  = cnt + 256 * 16;          // 16
  float* mrow = (float*)(dig + 16);    // 4096

  const int q = blockIdx.x;
  const int t = threadIdx.x;

  // Load + order-preserving float->uint transform.
#pragma unroll
  for (int e = 0; e < 16; ++e) {
    int j = t + e * 256;
    u32 u = __float_as_uint(score[(long)q * kS + j]);
    u = (u >> 31) ? ~u : (u | 0x80000000u);
    keyA[j] = u;
    idxA[j] = (u32)j;
  }
  __syncthreads();

  u32 *ki = keyA, *ko = keyB, *ii = idxA, *io = idxB;
  for (int p = 0; p < 8; ++p) {
    const int sh = p * 4;
#pragma unroll
    for (int d = 0; d < 16; ++d) cnt[t * 16 + d] = 0;
    for (int e = 0; e < 16; ++e) {
      u32 d = (ki[t * 16 + e] >> sh) & 15u;
      cnt[t * 16 + d] += 1;
    }
    __syncthreads();
    if (t < 16) {                       // per-digit scan across thread chunks
      u32 run = 0;
      for (int tt = 0; tt < 256; ++tt) {
        u32 c = cnt[tt * 16 + t];
        cnt[tt * 16 + t] = run;
        run += c;
      }
      dig[t] = run;
    }
    __syncthreads();
    if (t == 0) {                       // exclusive scan of digit totals
      u32 run = 0;
      for (int d = 0; d < 16; ++d) { u32 c = dig[d]; dig[d] = run; run += c; }
    }
    __syncthreads();
    for (int e = 0; e < 16; ++e) {      // stable scatter
      u32 u = ki[t * 16 + e];
      u32 ix = ii[t * 16 + e];
      u32 d = (u >> sh) & 15u;
      u32 pos = dig[d] + cnt[t * 16 + d];
      cnt[t * 16 + d] += 1;
      ko[pos] = u;
      io[pos] = ix;
    }
    __syncthreads();
    u32* tk = ki; ki = ko; ko = tk;
    u32* ti = ii; ii = io; io = ti;
  }
  // ii[] now index-sorted ascending by key; j-th largest at 4095-j.

#pragma unroll
  for (int e = 0; e < 8; ++e) {
    int j = t + e * 256;
    int iv = (int)ii[kS - 1 - j];
    if (q < j) iv = kS;                 // "future" replacement
    topk[(long)q * kTOPK + j] = iv;
  }

#pragma unroll
  for (int e = 0; e < 16; ++e) mrow[t + e * 256] = kMIN;
  __syncthreads();
  const int lim = (q < kTOPK - 1) ? q : (kTOPK - 1);
  for (int j = t; j <= lim; j += 256) mrow[ii[kS - 1 - j]] = 0.0f;
  __syncthreads();

#pragma unroll
  for (int e = 0; e < 16; ++e) {
    int k = t + e * 256;
    float m = mrow[k];
    maskout[(long)q * kS + k] = m;
    float raw = score[(long)q * kS + k];
    score[(long)q * kS + k] = fmaxf(raw + m, kMIN);
  }
}

// ---------------------------------------------------------------------------

extern "C" void kernel_launch(void* const* d_in, const int* in_sizes, int n_in,
                              void* d_out, int out_size, void* d_ws, size_t ws_size,
                              hipStream_t stream) {
  (void)in_sizes; (void)n_in; (void)out_size; (void)ws_size;

  const float* x    = (const float*)d_in[0];
  const float* qr   = (const float*)d_in[1];
  const float* cosp = (const float*)d_in[2];
  const float* sinp = (const float*)d_in[3];
  /* d_in[4] = causal mask: recomputed analytically */
  const float* wq_b = (const float*)d_in[5];
  const float* wk   = (const float*)d_in[6];
  const float* knw  = (const float*)d_in[7];
  const float* ww   = (const float*)d_in[8];

  char* ws = (char*)d_ws;
  __bf16* xb   = (__bf16*)ws; ws += (size_t)kS * kDIM * 2;        // 24 MB
  __bf16* qrb  = (__bf16*)ws; ws += (size_t)kS * kQLORA * 2;      // 12 MB
  __bf16* wqbT = (__bf16*)ws; ws += (size_t)(kH * kD) * kQLORA * 2; // 6 MB
  __bf16* wkT  = (__bf16*)ws; ws += (size_t)kD * kDIM * 2;        // 768 KB
  __bf16* wwT  = (__bf16*)ws; ws += (size_t)kH * kDIM * 2;        // 96 KB
  __bf16* qfb  = (__bf16*)ws; ws += (size_t)kH * kS * kD * 2;     // 16 MB
  __bf16* kfb  = (__bf16*)ws; ws += (size_t)kS * kD * 2;          // 1 MB
  float*  wts  = (float*)ws;  ws += (size_t)kS * kH * 4;          // 256 KB

  int*   topk    = (int*)d_out;
  float* score   = (float*)d_out + (size_t)kS * kTOPK;
  float* maskout = score + (size_t)kS * kS;

  // k0: bf16 staging
  cvt_bf16_kernel<<<dim3(kS * kDIM / 256), 256, 0, stream>>>(x, xb, kS * kDIM);
  cvt_bf16_kernel<<<dim3(kS * kQLORA / 256), 256, 0, stream>>>(qr, qrb, kS * kQLORA);
  transpose_cvt_kernel<<<dim3((kH * kD) / 32, kQLORA / 32), dim3(32, 8), 0, stream>>>(
      wq_b, wqbT, kQLORA, kH * kD);
  transpose_cvt_kernel<<<dim3(kD / 32, kDIM / 32), dim3(32, 8), 0, stream>>>(
      wk, wkT, kDIM, kD);
  transpose_cvt_kernel<<<dim3(1, kDIM / 32), dim3(32, 8), 0, stream>>>(
      ww, wwT, kDIM, kH);

  // k1: q projection + RoPE
  qproj_kernel<<<dim3(kS / 16, kH), 256, 0, stream>>>(qrb, wqbT, cosp, sinp, qfb);

  // k2: k projection + rmsnorm + RoPE + weights
  kproj_kernel<<<dim3(kS / 16), 288, 0, stream>>>(xb, wkT, wwT, cosp, sinp, knw,
                                                  kfb, wts);

  // k3: scores + head reduction + causal mask (TDM-staged q tile)
  const size_t smem3 = (size_t)kH * 16 * kD * 2 + 16 * kH * 4;   // 65 KB
  scores_kernel<<<dim3(kS / 16, kS / 128), 256, smem3, stream>>>(qfb, kfb, wts, score);

  // k4: per-row top-k + masks + final clamp
  const size_t smem4 = (size_t)4 * kS * 4 + 256 * 16 * 4 + 16 * 4 + kS * 4; // ~96 KB
  topk_kernel<<<dim3(kS), 256, smem4, stream>>>(score, topk, maskout);
}
